// MPSv3_14130442404278
// MI455X (gfx1250) — matrix operand order repaired and checked by reading, not compiled, not run on hardware
//
#include <hip/hip_runtime.h>

typedef __attribute__((ext_vector_type(16))) _Float16 v16h;
typedef __attribute__((ext_vector_type(8)))  _Float16 v8h;
typedef __attribute__((ext_vector_type(8)))  float    v8f;

namespace {
constexpr int IN_DIM  = 512;
constexpr int NEURONS = 1024;
constexpr int OUT_DIM = 512;
constexpr int NBASIS  = 18;     // NUM_KNOTS + DEGREE - 1
constexpr int BM      = 32;     // batch rows per workgroup (2 row-tiles)
constexpr int NTHR    = 512;    // 16 wave32s
constexpr int XSTR    = 520;    // halfs: x staging stride (16B-chunk aligned, bank spread)
constexpr int HSTR    = 1040;   // halfs: h / spline buffer stride (multiple of 8)
}

union FragU { v16h v; v8h h[2]; };

// A-matrix 16x32 f16 layout: lane L holds row M=L%16, K-chunks at
// (L/16)*8..+7 and 16+(L/16)*8..+7 (two 16B loads).
__device__ __forceinline__ v16h frag_a(const _Float16* base) {
  FragU f;
  f.h[0] = *(const v8h*)(base);
  f.h[1] = *(const v8h*)(base + 16);
  return f.v;
}
// B-matrix 32x16 f16 layout: lane L holds column N=L%16, K-chunks at
// (L/16)*8..+7 and 16+(L/16)*8..+7 (two 16B loads).
__device__ __forceinline__ v16h frag_b(const _Float16* base) {
  FragU f;
  f.h[0] = *(const v8h*)(base);
  f.h[1] = *(const v8h*)(base + 16);
  return f.v;
}
__device__ __forceinline__ v8f zero8() {
  v8f z = {0.f,0.f,0.f,0.f,0.f,0.f,0.f,0.f};
  return z;
}

// Convert W1 (512x1024) -> W1^T f16 [n][k], W2 (1024x512) -> W2^T f16 [o][n].
extern "C" __global__ __launch_bounds__(256)
void kan_convert(const float* __restrict__ W1, const float* __restrict__ W2,
                 _Float16* __restrict__ w1t, _Float16* __restrict__ w2t) {
  const int idx = blockIdx.x * 256 + threadIdx.x;   // 0 .. 524287
  {
    const int k = idx & (IN_DIM - 1);
    const int n = idx >> 9;
    w1t[(size_t)n * IN_DIM + k] = (_Float16)W1[(size_t)k * NEURONS + n];
  }
  {
    const int n = idx & (NEURONS - 1);
    const int o = idx >> 10;
    w2t[(size_t)o * NEURONS + n] = (_Float16)W2[(size_t)n * OUT_DIM + o];
  }
}

extern "C" __global__ __launch_bounds__(NTHR)
void kan_fused(const float* __restrict__ x,
               const _Float16* __restrict__ w1t,
               const float* __restrict__ b1,
               const float* __restrict__ coeff,
               const float* __restrict__ sp_bias,
               const _Float16* __restrict__ w2t,
               const float* __restrict__ b2,
               float* __restrict__ out) {
  __shared__ _Float16 xh[BM * XSTR];   // 33.3 KB : x rows, f16
  __shared__ _Float16 hb[BM * HSTR];   // 66.6 KB : h, then (in place) spline acts
  __shared__ float    pmin[NTHR], pmax[NTHR];
  __shared__ float    rmin[BM], rmax[BM];

  const int tid  = threadIdx.x;
  const int wave = tid >> 5;           // 0..15
  const int lane = tid & 31;
  const int lr   = lane & 15;
  const int ka   = (lane >> 4) * 8;    // A-frag K sub-offset
  const int kb   = (lane >> 4) * 8;    // B-frag K sub-offset
  const int m0   = blockIdx.x * BM;

  // ---- phase 1: stage x rows into LDS as f16 (coalesced) ----
  for (int i = tid; i < BM * IN_DIM; i += NTHR) {
    const int r = i >> 9, k = i & (IN_DIM - 1);
    xh[r * XSTR + k] = (_Float16)x[(size_t)(m0 + r) * IN_DIM + k];
  }
  __syncthreads();

  // ---- phase 2: h = x @ W1 + b1 ----
  // Each wave: 2 row-tiles x 4 col-tiles; each B fragment feeds 2 WMMAs.
  {
    v8f acc[2][4];
#pragma unroll
    for (int rt = 0; rt < 2; ++rt)
#pragma unroll
      for (int t = 0; t < 4; ++t) acc[rt][t] = zero8();
    const int ct0 = wave * 4;          // col tiles [ct0, ct0+3]
    for (int k0 = 0; k0 < IN_DIM; k0 += 32) {
      const v16h a0 = frag_a(&xh[lr * XSTR + k0 + ka]);
      const v16h a1 = frag_a(&xh[(16 + lr) * XSTR + k0 + ka]);
#pragma unroll
      for (int t = 0; t < 4; ++t) {
        const int n = (ct0 + t) * 16 + lr;
        const v16h b = frag_b(w1t + (size_t)n * IN_DIM + k0 + kb);
        acc[0][t] = __builtin_amdgcn_wmma_f32_16x16x32_f16(
            false, a0, false, b, (short)0, acc[0][t], false, false);
        acc[1][t] = __builtin_amdgcn_wmma_f32_16x16x32_f16(
            false, a1, false, b, (short)0, acc[1][t], false, false);
      }
    }
#pragma unroll
    for (int rt = 0; rt < 2; ++rt) {
#pragma unroll
      for (int t = 0; t < 4; ++t) {
        const int n = (ct0 + t) * 16 + lr;
        const float bias = b1[n];
#pragma unroll
        for (int i = 0; i < 8; ++i) {
          const int row = rt * 16 + ((lane < 16) ? i : (8 + i));
          hb[row * HSTR + n] = (_Float16)(acc[rt][t][i] + bias);
        }
      }
    }
  }
  __syncthreads();

  // ---- phase 3: per-row min / max over 1024 neurons ----
  {
    const int r = tid & 31, seg = tid >> 5;          // 16 segments of 64
    const _Float16* hp = &hb[r * HSTR + seg * 64];
    float mn = 3.402823466e38f, mx = -3.402823466e38f;
    for (int i = 0; i < 64; ++i) {
      const float v = (float)hp[i];
      mn = fminf(mn, v);
      mx = fmaxf(mx, v);
    }
    pmin[seg * BM + r] = mn;
    pmax[seg * BM + r] = mx;
  }
  __syncthreads();
  if (tid < BM) {
    float mn = pmin[tid], mx = pmax[tid];
    for (int s = 1; s < 16; ++s) {
      mn = fminf(mn, pmin[s * BM + tid]);
      mx = fmaxf(mx, pmax[s * BM + tid]);
    }
    rmin[tid] = mn;
    rmax[tid] = mx;
  }
  __syncthreads();

  // ---- phase 4: normalize + uniform-knot cubic B-spline + coeff dot,
  //               written back IN PLACE over h in LDS ----
  for (int i = tid; i < BM * NEURONS; i += NTHR) {
    const int r = i >> 10, n = i & (NEURONS - 1);
    const float mn  = rmin[r];
    const float inv = 1.0f / (rmax[r] - mn + 1e-8f);
    const float u   = ((float)hb[r * HSTR + n] - mn) * inv;   // in [0,1)
    const float s   = u * 15.0f;                              // 15 = NUM_KNOTS-1
    int cell = (int)s;
    cell = cell < 0 ? 0 : (cell > 14 ? 14 : cell);
    const float t   = s - (float)cell;
    const float omt = 1.0f - t;
    const float t2  = t * t, t3 = t2 * t;
    const float k6  = 1.0f / 6.0f;
    const float w0  = omt * omt * omt * k6;
    const float w1  = (3.0f * t3 - 6.0f * t2 + 4.0f) * k6;
    const float w2  = (-3.0f * t3 + 3.0f * t2 + 3.0f * t + 1.0f) * k6;
    const float w3  = t3 * k6;
    const float* c  = coeff + (size_t)n * NBASIS + cell;
    const float sp  = w0 * c[0] + w1 * c[1] + w2 * c[2] + w3 * c[3] + sp_bias[n];
    hb[r * HSTR + n] = (_Float16)sp;
  }
  __syncthreads();

  // ---- phase 5: out = relu(sp @ W2 + b2) ----
  // Each wave: 2 row-tiles x 2 col-tiles; each B fragment feeds 2 WMMAs.
  {
    v8f acc[2][2];
#pragma unroll
    for (int rt = 0; rt < 2; ++rt)
#pragma unroll
      for (int t = 0; t < 2; ++t) acc[rt][t] = zero8();
    const int ct0 = wave * 2;          // 32 col tiles / 16 waves
    for (int k0 = 0; k0 < NEURONS; k0 += 32) {
      const v16h a0 = frag_a(&hb[lr * HSTR + k0 + ka]);
      const v16h a1 = frag_a(&hb[(16 + lr) * HSTR + k0 + ka]);
#pragma unroll
      for (int t = 0; t < 2; ++t) {
        const int o = (ct0 + t) * 16 + lr;
        const v16h b = frag_b(w2t + (size_t)o * NEURONS + k0 + kb);
        acc[0][t] = __builtin_amdgcn_wmma_f32_16x16x32_f16(
            false, a0, false, b, (short)0, acc[0][t], false, false);
        acc[1][t] = __builtin_amdgcn_wmma_f32_16x16x32_f16(
            false, a1, false, b, (short)0, acc[1][t], false, false);
      }
    }
#pragma unroll
    for (int rt = 0; rt < 2; ++rt) {
#pragma unroll
      for (int t = 0; t < 2; ++t) {
        const int o = (ct0 + t) * 16 + lr;
        const float bias = b2[o];
#pragma unroll
        for (int i = 0; i < 8; ++i) {
          const int row = rt * 16 + ((lane < 16) ? i : (8 + i));
          const float v = acc[rt][t][i] + bias;
          out[(size_t)(m0 + row) * OUT_DIM + o] = fmaxf(v, 0.0f);
        }
      }
    }
  }
}

extern "C" void kernel_launch(void* const* d_in, const int* in_sizes, int n_in,
                              void* d_out, int out_size, void* d_ws, size_t ws_size,
                              hipStream_t stream) {
  (void)in_sizes; (void)n_in; (void)out_size; (void)ws_size;
  const float* x       = (const float*)d_in[0];
  const float* W1      = (const float*)d_in[1];
  const float* b1      = (const float*)d_in[2];
  const float* coeff   = (const float*)d_in[3];
  const float* sp_bias = (const float*)d_in[4];
  const float* W2      = (const float*)d_in[5];
  const float* b2      = (const float*)d_in[6];

  _Float16* w1t = (_Float16*)d_ws;                        // 1024*512 halfs = 1 MB
  _Float16* w2t = w1t + (size_t)NEURONS * IN_DIM;         // 512*1024 halfs = 1 MB

  kan_convert<<<2048, 256, 0, stream>>>(W1, W2, w1t, w2t);
  kan_fused<<<4096 / BM, NTHR, 0, stream>>>(x, w1t, b1, coeff, sp_bias, w2t, b2,
                                            (float*)d_out);
}
